// CGC_73675868996392
// MI455X (gfx1250) — matrix operand order repaired and checked by reading, not compiled, not run on hardware
//
#include <hip/hip_runtime.h>
#include <hip/hip_bf16.h>

// CGC / multi-gate MoE, fully fused for MI455X (gfx1250, wave32, WMMA).
//
// B=65536, I=512, H=128, T=2, ES=2, EC=4, ETOT=6.
//
// Fast path (needs 2MB workspace):
//   kernel 1: pack fp32 weights -> split-precision f16 hi/lo, pre-swizzled into
//             a 32x16 f16 B-fragment layout with contiguous k (k = 16h + s), hi/lo interleaved so each
//             lane's (hi,lo) fragment pair is 64 contiguous bytes.
//   kernel 2: per 32-row tile of B (2048 workgroups):
//     phase 1 : stage x tile into LDS as f16 hi + f16 lo (split precision)
//     phase 2 : wave w computes expert w's 32x128 tile with
//               V_WMMA_F32_16X16X32_F16, 3 passes (hi*hi + lo*hi + hi*lo),
//               two M-tiles sharing every B fragment (halves L2 weight traffic,
//               0.67 b128 loads per WMMA).
//     phase 2b: + bias, spill 8x32x128 expert block to LDS (overlays x tile)
//     phase 3 : gate logits, softmax, weighted feature, Wt dot
// Fallback path: f32 V_WMMA_F32_16X16X4_F32 kernel (no workspace).

#define B_TOT   65536
#define I_DIM   512
#define H_DIM   128
#define T_N     2
#define ES_N    2
#define EC_N    4
#define ETOT_N  6
#define NEXP    8            // EC + T*ES distinct GEMMs
#define ROWS    16           // rows per WG (fallback kernel)
#define ROWS2   32           // rows per WG (fast kernel)
#define XS      516          // padded LDS stride, f32 x tile (floats)
#define XSH     520          // padded LDS stride, f16 x tiles (halfs)
#define EPS     132          // padded LDS stride, expert rows (floats)
#define NKB     (I_DIM / 32) // 16 K-blocks of 32
#define NTILES  (H_DIM / 16) // 8 N-tiles of 16
#define FRAG_P  1024         // halfs per packed (hi+lo) fragment pair (32 lanes x 32)

typedef __attribute__((ext_vector_type(2)))  float     v2f;
typedef __attribute__((ext_vector_type(8)))  float     v8f;
typedef __attribute__((ext_vector_type(4)))  _Float16  v4h;
typedef __attribute__((ext_vector_type(8)))  _Float16  v8h;
typedef __attribute__((ext_vector_type(16))) _Float16  v16h;

// ---------------------------------------------------------------------------
// Weight pre-pack: fp32 -> f16 hi/lo, swizzled to B-fragment layout.
// Fragment pair (e, kb, nt): 32 lanes x (16 hi + 16 lo) halfs.
// Lane ln (h=ln>>4, n=ln&15): hi dword j = pair( W[kb*32+16h+2j][nt*16+n],
//                                               W[kb*32+16h+2j+1][nt*16+n] )
// lo follows at +16 halfs, so one lane's hi+lo = 64 contiguous bytes.
// ---------------------------------------------------------------------------
__global__ __launch_bounds__(256) void cgc_pack_weights(
    const float* __restrict__ Wc, const float* __restrict__ Ws,
    _Float16* __restrict__ pk)
{
    int gid  = blockIdx.x * 256 + threadIdx.x;    // 8*16*8*32 = 32768 total
    int lane = gid & 31;
    int nt   = (gid >> 5) & 7;
    int kb   = (gid >> 8) & 15;
    int e    = gid >> 12;
    const float* W = (e < EC_N) ? (Wc + (size_t)e * I_DIM * H_DIM)
                                : (Ws + (size_t)(e - EC_N) * I_DIM * H_DIM);
    int h  = lane >> 4;
    int n  = (lane & 15) + nt * 16;
    int k0 = kb * 32 + 16 * h;
    size_t base = ((size_t)(e * NKB + kb) * NTILES + nt) * FRAG_P + (size_t)lane * 32;
#pragma unroll
    for (int j = 0; j < 8; ++j) {
        float w0 = W[(size_t)(k0 + 2 * j)     * H_DIM + n];
        float w1 = W[(size_t)(k0 + 2 * j + 1) * H_DIM + n];
        _Float16 h0 = (_Float16)w0, h1 = (_Float16)w1;
        pk[base + 2 * j]          = h0;
        pk[base + 2 * j + 1]      = h1;
        pk[base + 16 + 2 * j]     = (_Float16)(w0 - (float)h0);
        pk[base + 16 + 2 * j + 1] = (_Float16)(w1 - (float)h1);
    }
}

// ---------------------------------------------------------------------------
// Fused main kernel: split-precision f16x3 WMMA, M=32 per wave.
// ---------------------------------------------------------------------------
__global__ __launch_bounds__(256) void cgc_fused_f16x3(
    const float* __restrict__ x, const _Float16* __restrict__ Wpk,
    const float* __restrict__ bc, const float* __restrict__ bs,
    const float* __restrict__ Wg, const float* __restrict__ bg,
    const float* __restrict__ Wt, const float* __restrict__ bt,
    float* __restrict__ out)
{
    __shared__ float smem[NEXP * ROWS2 * EPS + T_N * ROWS2 * ETOT_N]; // ~134KB
    _Float16* xhi = (_Float16*)smem;              // 32 x 520 halfs
    _Float16* xlo = xhi + ROWS2 * XSH;            // 32 x 520 halfs (66560B total)
    float* Ebuf = smem;                           // 8 x 32 x 132 f32, overlays x tiles
    float* lgt  = smem + NEXP * ROWS2 * EPS;      // [T][32][6] logits

    const int tid  = threadIdx.x;
    const int wave = tid >> 5;                    // wave32
    const int lane = tid & 31;
    const int half = lane >> 4;
    const int lid  = lane & 15;
    const int rowBase = blockIdx.x * ROWS2;

    // ---- Phase 1: stage x tile, split into f16 hi/lo in LDS ----------------
    {
        const float* xg = x + (size_t)rowBase * I_DIM;
#pragma unroll
        for (int j = 0; j < 16; ++j) {
            int i4 = tid + j * 256;               // 0..4095 float4s
            int r  = i4 >> 7;
            int c4 = i4 & 127;
            float4 v = *(const float4*)(xg + (size_t)r * I_DIM + c4 * 4);
            v4h hv, lv;
            hv[0] = (_Float16)v.x; lv[0] = (_Float16)(v.x - (float)hv[0]);
            hv[1] = (_Float16)v.y; lv[1] = (_Float16)(v.y - (float)hv[1]);
            hv[2] = (_Float16)v.z; lv[2] = (_Float16)(v.z - (float)hv[2]);
            hv[3] = (_Float16)v.w; lv[3] = (_Float16)(v.w - (float)hv[3]);
            *(v4h*)(xhi + r * XSH + c4 * 4) = hv;
            *(v4h*)(xlo + r * XSH + c4 * 4) = lv;
        }
    }
    __syncthreads();

    // ---- Phase 2: per-wave expert GEMM (32x512 x 512x128), f16x3 -----------
    const float* bp = (wave < EC_N) ? (bc + wave * H_DIM)
                                    : (bs + (wave - EC_N) * H_DIM);
    const _Float16* pkBase = Wpk + (size_t)wave * NKB * NTILES * FRAG_P
                                 + (size_t)lane * 32;

    v8f acc0[NTILES] = {};                        // rows 0..15
    v8f acc1[NTILES] = {};                        // rows 16..31

#pragma unroll 2
    for (int kb = 0; kb < NKB; ++kb) {
        const int k0 = kb * 32;
        // A fragments (16x32 f16 layout): lane row = lid (+16 for 2nd M tile);
        // VGPR0-7 = K[k0+16h .. +15]
        const _Float16* a0h = xhi + lid * XSH + k0 + 16 * half;
        const _Float16* a0l = xlo + lid * XSH + k0 + 16 * half;
        const _Float16* a1h = a0h + 16 * XSH;
        const _Float16* a1l = a0l + 16 * XSH;
        v16h aHi0 = __builtin_shufflevector(*(const v8h*)a0h, *(const v8h*)(a0h + 8),
                                            0,1,2,3,4,5,6,7,8,9,10,11,12,13,14,15);
        v16h aLo0 = __builtin_shufflevector(*(const v8h*)a0l, *(const v8h*)(a0l + 8),
                                            0,1,2,3,4,5,6,7,8,9,10,11,12,13,14,15);
        v16h aHi1 = __builtin_shufflevector(*(const v8h*)a1h, *(const v8h*)(a1h + 8),
                                            0,1,2,3,4,5,6,7,8,9,10,11,12,13,14,15);
        v16h aLo1 = __builtin_shufflevector(*(const v8h*)a1l, *(const v8h*)(a1l + 8),
                                            0,1,2,3,4,5,6,7,8,9,10,11,12,13,14,15);

        const _Float16* fb = pkBase + (size_t)kb * NTILES * FRAG_P;
        __builtin_prefetch(fb + NTILES * FRAG_P, 0, 0);   // next K-block (16KB ahead)
#pragma unroll
        for (int nt = 0; nt < NTILES; ++nt) {
            const _Float16* f = fb + nt * FRAG_P;
            v16h bHi = *(const v16h*)f;           // 32B contiguous per lane
            v16h bLo = *(const v16h*)(f + 16);
            acc0[nt] = __builtin_amdgcn_wmma_f32_16x16x32_f16(
                false, aHi0, false, bHi, (short)0, acc0[nt], false, false);
            acc1[nt] = __builtin_amdgcn_wmma_f32_16x16x32_f16(
                false, aHi1, false, bHi, (short)0, acc1[nt], false, false);
            acc0[nt] = __builtin_amdgcn_wmma_f32_16x16x32_f16(
                false, aLo0, false, bHi, (short)0, acc0[nt], false, false);
            acc1[nt] = __builtin_amdgcn_wmma_f32_16x16x32_f16(
                false, aLo1, false, bHi, (short)0, acc1[nt], false, false);
            acc0[nt] = __builtin_amdgcn_wmma_f32_16x16x32_f16(
                false, aHi0, false, bLo, (short)0, acc0[nt], false, false);
            acc1[nt] = __builtin_amdgcn_wmma_f32_16x16x32_f16(
                false, aHi1, false, bLo, (short)0, acc1[nt], false, false);
        }
    }

    __syncthreads();   // all waves done reading x tiles before Ebuf overlays them

    // ---- Phase 2b: bias + spill expert tile to LDS -------------------------
#pragma unroll
    for (int nt = 0; nt < NTILES; ++nt) {
        float be = bp[nt * 16 + lid];
#pragma unroll
        for (int v = 0; v < 8; ++v) {
            int r = v + half * 8;                 // C layout: VGPR v -> row v + 8*half
            Ebuf[(wave * ROWS2 + r) * EPS + nt * 16 + lid]      = acc0[nt][v] + be;
            Ebuf[(wave * ROWS2 + 16 + r) * EPS + nt * 16 + lid] = acc1[nt][v] + be;
        }
    }
    __syncthreads();

    // ---- Phase 3a: gate logits [T][32][6] ----------------------------------
    for (int it = tid; it < T_N * ROWS2 * ETOT_N; it += 256) {   // 384 jobs
        int t  = it / (ROWS2 * ETOT_N);
        int rr = (it / ETOT_N) % ROWS2;
        int e  = it % ETOT_N;
        float aL = bg[t * ETOT_N + e];
        const float* wg = Wg + (size_t)t * (ETOT_N * H_DIM) * ETOT_N + e;
#pragma unroll
        for (int e2 = 0; e2 < ETOT_N; ++e2) {
            int slot = (e2 < ES_N) ? (EC_N + t * ES_N + e2) : (e2 - ES_N);
            const float* er  = Ebuf + (slot * ROWS2 + rr) * EPS;
            const float* wgf = wg + (size_t)(e2 * H_DIM) * ETOT_N;
            float s = 0.f;
#pragma unroll 4
            for (int h = 0; h < H_DIM; ++h) s += er[h] * wgf[(size_t)h * ETOT_N];
            aL += s;
        }
        lgt[it] = aL;
    }
    __syncthreads();

    // ---- Phase 3b: softmax, weighted feature, output -----------------------
    if (tid < T_N * ROWS2) {                      // 64 threads: (t, row)
        int t  = tid >> 5;
        int rr = tid & 31;
        const float* l = lgt + (t * ROWS2 + rr) * ETOT_N;
        float m = l[0];
#pragma unroll
        for (int e = 1; e < ETOT_N; ++e) m = fmaxf(m, l[e]);
        float w[ETOT_N];
        float sum = 0.f;
#pragma unroll
        for (int e = 0; e < ETOT_N; ++e) { w[e] = __expf(l[e] - m); sum += w[e]; }
        float inv = 1.f / sum;
        float o = bt[t];
#pragma unroll
        for (int e = 0; e < ETOT_N; ++e) {
            int slot = (e < ES_N) ? (EC_N + t * ES_N + e) : (e - ES_N);
            const float* er = Ebuf + (slot * ROWS2 + rr) * EPS;
            float s = 0.f;
#pragma unroll 4
            for (int h = 0; h < H_DIM; ++h) s += er[h] * Wt[t * H_DIM + h];
            o += (w[e] * inv) * s;
        }
        out[(size_t)t * B_TOT + rowBase + rr] = o;
    }
}

// ---------------------------------------------------------------------------
// Fallback: f32 V_WMMA_F32_16X16X4_F32 kernel (no workspace needed).
// ---------------------------------------------------------------------------
__global__ __launch_bounds__(256) void cgc_fused_f32(
    const float* __restrict__ x,  const float* __restrict__ Wc,
    const float* __restrict__ bc, const float* __restrict__ Ws,
    const float* __restrict__ bs, const float* __restrict__ Wg,
    const float* __restrict__ bg, const float* __restrict__ Wt,
    const float* __restrict__ bt, float* __restrict__ out)
{
    __shared__ float smem[NEXP * ROWS * EPS + T_N * ROWS * ETOT_N];
    float* xs   = smem;
    float* Ebuf = smem;
    float* lgt  = smem + NEXP * ROWS * EPS;

    const int tid  = threadIdx.x;
    const int wave = tid >> 5;
    const int lane = tid & 31;
    const int half = lane >> 4;
    const int lid  = lane & 15;
    const int rowBase = blockIdx.x * ROWS;

    {
        const float* xg = x + (size_t)rowBase * I_DIM;
#pragma unroll
        for (int j = 0; j < 8; ++j) {
            int i4 = tid + j * 256;
            int r  = i4 >> 7;
            int c4 = i4 & 127;
            float4 v = *(const float4*)(xg + (size_t)r * I_DIM + c4 * 4);
            *(float4*)(xs + r * XS + c4 * 4) = v;
        }
    }
    __syncthreads();

    const float* Wp = (wave < EC_N) ? (Wc + (size_t)wave * I_DIM * H_DIM)
                                    : (Ws + (size_t)(wave - EC_N) * I_DIM * H_DIM);
    const float* bp = (wave < EC_N) ? (bc + wave * H_DIM)
                                    : (bs + (wave - EC_N) * H_DIM);

    v8f acc[8] = {};
    const int kA0 = half * 2;
#pragma unroll 2
    for (int kb = 0; kb < I_DIM / 4; ++kb) {
        const int kA = kb * 4 + kA0;
        v2f a = *(const v2f*)(xs + lid * XS + kA);
        const float* wrow0 = Wp + (size_t)kA * H_DIM;
        const float* wrow1 = wrow0 + H_DIM;
        __builtin_prefetch(wrow0 + 16 * H_DIM, 0, 0);
#pragma unroll
        for (int nt = 0; nt < 8; ++nt) {
            v2f b;
            b.x = wrow0[nt * 16 + lid];
            b.y = wrow1[nt * 16 + lid];
            acc[nt] = __builtin_amdgcn_wmma_f32_16x16x4_f32(
                false, a, false, b, (short)0, acc[nt], false, false);
        }
    }
    __syncthreads();

#pragma unroll
    for (int nt = 0; nt < 8; ++nt) {
        float be = bp[nt * 16 + lid];
#pragma unroll
        for (int v = 0; v < 8; ++v) {
            int r = v + half * 8;
            Ebuf[(wave * ROWS + r) * EPS + nt * 16 + lid] = acc[nt][v] + be;
        }
    }
    __syncthreads();

    if (tid < T_N * ROWS * ETOT_N) {
        int t  = tid / (ROWS * ETOT_N);
        int rr = (tid / ETOT_N) % ROWS;
        int e  = tid % ETOT_N;
        float aL = bg[t * ETOT_N + e];
        const float* wg = Wg + (size_t)t * (ETOT_N * H_DIM) * ETOT_N + e;
#pragma unroll
        for (int e2 = 0; e2 < ETOT_N; ++e2) {
            int slot = (e2 < ES_N) ? (EC_N + t * ES_N + e2) : (e2 - ES_N);
            const float* er  = Ebuf + (slot * ROWS + rr) * EPS;
            const float* wgf = wg + (size_t)(e2 * H_DIM) * ETOT_N;
            float s = 0.f;
#pragma unroll 4
            for (int h = 0; h < H_DIM; ++h) s += er[h] * wgf[(size_t)h * ETOT_N];
            aL += s;
        }
        lgt[tid] = aL;
    }
    __syncthreads();

    if (tid < T_N * ROWS) {
        int t  = tid >> 4;
        int rr = tid & 15;
        const float* l = lgt + (t * ROWS + rr) * ETOT_N;
        float m = l[0];
#pragma unroll
        for (int e = 1; e < ETOT_N; ++e) m = fmaxf(m, l[e]);
        float w[ETOT_N];
        float sum = 0.f;
#pragma unroll
        for (int e = 0; e < ETOT_N; ++e) { w[e] = __expf(l[e] - m); sum += w[e]; }
        float inv = 1.f / sum;
        float o = bt[t];
#pragma unroll
        for (int e = 0; e < ETOT_N; ++e) {
            int slot = (e < ES_N) ? (EC_N + t * ES_N + e) : (e - ES_N);
            const float* er = Ebuf + (slot * ROWS + rr) * EPS;
            float s = 0.f;
#pragma unroll 4
            for (int h = 0; h < H_DIM; ++h) s += er[h] * Wt[t * H_DIM + h];
            o += (w[e] * inv) * s;
        }
        out[(size_t)t * B_TOT + rowBase + rr] = o;
    }
}

extern "C" void kernel_launch(void* const* d_in, const int* in_sizes, int n_in,
                              void* d_out, int out_size, void* d_ws, size_t ws_size,
                              hipStream_t stream) {
    (void)in_sizes; (void)n_in; (void)out_size;
    const float* x  = (const float*)d_in[0];
    const float* Wc = (const float*)d_in[1];
    const float* bc = (const float*)d_in[2];
    const float* Ws = (const float*)d_in[3];
    const float* bs = (const float*)d_in[4];
    const float* Wg = (const float*)d_in[5];
    const float* bg = (const float*)d_in[6];
    const float* Wt = (const float*)d_in[7];
    const float* bt = (const float*)d_in[8];
    float* out = (float*)d_out;

    const size_t need = (size_t)NEXP * I_DIM * H_DIM * 2 * sizeof(_Float16); // 2 MB

    if (ws_size >= need) {
        _Float16* pk = (_Float16*)d_ws;
        cgc_pack_weights<<<dim3(128), dim3(256), 0, stream>>>(Wc, Ws, pk);
        cgc_fused_f16x3<<<dim3(B_TOT / ROWS2), dim3(256), 0, stream>>>(
            x, pk, bc, bs, Wg, bg, Wt, bt, out);
    } else {
        cgc_fused_f32<<<dim3(B_TOT / ROWS), dim3(256), 0, stream>>>(
            x, Wc, bc, Ws, bs, Wg, bg, Wt, bt, out);
    }
}
